// SelfAttention_38062000177790
// MI455X (gfx1250) — compile-verified, matrix operand order checked by reading, not run on hardware
//
#include <hip/hip_runtime.h>
#include <hip/hip_bf16.h>

// ---------------------------------------------------------------------------
// Self-attention for MI455X (gfx1250, wave32, WMMA).
//   H [B,D,L] f32 -> K,Q = ELU(W*H+b) [b,l,d] bf16 ; V stored transposed [b,d,l]
//   E = Q K^T / sqrt(512) -> d_out A region (f32), 2x2 tiles/wave
//   softmax rows in place -> A f32 (d_out) + A bf16 (ws)
//   C = A V  (via V^T)    -> d_out C region (f32), 2x2 tiles/wave
// All matmuls use v_wmma_f32_16x16x32_bf16 (f32 accumulate).
// ---------------------------------------------------------------------------

typedef __attribute__((ext_vector_type(16))) __bf16 v16bf;
typedef __attribute__((ext_vector_type(8)))  __bf16 v8bf;
typedef __attribute__((ext_vector_type(8)))  float  v8f;

constexpr int B = 8, D = 512, L = 2048;
constexpr float SCALE_E = 0.044194173824159216f; // 1/sqrt(512)
constexpr size_t BLD = (size_t)B * L * D;        // 8,388,608 elems
constexpr size_t BLL = (size_t)B * L * L;        // 33,554,432 elems

__device__ __forceinline__ v8f wmma_bf16(v16bf a, v16bf b, v8f c) {
  return __builtin_amdgcn_wmma_f32_16x16x32_bf16(
      /*neg_a=*/false, a, /*neg_b=*/false, b,
      /*c_mod=*/(short)0, c, /*reuse_a=*/false, /*reuse_b=*/false);
}

// Load a 16-bit WMMA fragment whose per-lane K-values are contiguous in
// memory: two 16-byte vector loads at +0 and +16 (elements).
__device__ __forceinline__ v16bf frag_ld(const __bf16* p) {
  v8bf lo = *(const v8bf*)(p);
  v8bf hi = *(const v8bf*)(p + 16);
  v16bf r;
#pragma unroll
  for (int e = 0; e < 8; ++e) { r[e] = lo[e]; r[8 + e] = hi[e]; }
  return r;
}

// ---------------------------------------------------------------------------
// Kernel 1: projections. One wave per 16x16 output tile of {K,Q,V}.
// K,Q stored [b,l,d]; V stored transposed [b,d,l] so the A*V GEMM gets
// contiguous-K fragments for both operands.
// A-frag: A[m][k] = H[b, k, lbase+m]  (strided over L)
// B-frag: B[k][n] = W[obase+n, k]     (contiguous over k)
// ---------------------------------------------------------------------------
__global__ __launch_bounds__(256) void proj_kernel(
    const float* __restrict__ H,
    const float* __restrict__ Wk, const float* __restrict__ bk,
    const float* __restrict__ Wq, const float* __restrict__ bq,
    const float* __restrict__ Wv, const float* __restrict__ bv,
    __bf16* __restrict__ KQV) {
  const int lane = threadIdx.x & 31;
  const int wave = threadIdx.x >> 5;
  const int t = blockIdx.x * 8 + wave;          // 3*B*128*32 = 98304 tiles
  const int which = t >> 15;                    // 0=K, 1=Q, 2=V
  const int rt = t & 32767;
  const int b  = rt >> 12;
  const int lt = (rt >> 5) & 127;
  const int ot = rt & 31;

  const float* W    = (which == 0) ? Wk : ((which == 1) ? Wq : Wv);
  const float* bias = (which == 0) ? bk : ((which == 1) ? bq : bv);
  __bf16* out = KQV + (size_t)which * BLD;

  const int lcol = lane & 15;
  const int kb   = (lane & 16) ? 8 : 0;         // fragment K base per lane half
  const int lbase = lt * 16, obase = ot * 16;

  const float* hp = H + (size_t)b * D * L + (size_t)(lbase + lcol);
  const float* wp = W + (size_t)(obase + lcol) * D + kb;

  v8f acc = {};
#pragma unroll 4
  for (int k0 = 0; k0 < D; k0 += 32) {
    v16bf a, bbm;
#pragma unroll
    for (int e = 0; e < 8; ++e) {
      a[e]     = (__bf16)hp[(size_t)(k0 + kb + e)      * L];
      a[8 + e] = (__bf16)hp[(size_t)(k0 + kb + 16 + e) * L];
      bbm[e]     = (__bf16)wp[k0 + e];
      bbm[8 + e] = (__bf16)wp[k0 + 16 + e];
    }
    acc = wmma_bf16(a, bbm, acc);
  }

  const float bval = bias[obase + lcol];        // output column fixed per lane
  const int moff = (lane & 16) ? 8 : 0;
#pragma unroll
  for (int r = 0; r < 8; ++r) {
    float v = acc[r] + bval;
    v = (v > 0.f) ? v : (__expf(v) - 1.f);      // ELU
    const int lrow = lbase + moff + r;          // l index
    const int ocol = obase + lcol;              // d index
    if (which == 2) {                            // V^T: [b, d, l]
      out[((size_t)b * D + ocol) * L + lrow] = (__bf16)v;
    } else {                                     // K/Q: [b, l, d]
      out[((size_t)b * L + lrow) * D + ocol] = (__bf16)v;
    }
  }
}

// ---------------------------------------------------------------------------
// Kernel 2: E = Q K^T * scale. One wave per 32x32 tile (2x2 WMMA tiles):
// 4 wmma per 8 b128 fragment loads. Writes f32 logits straight into the
// A-output region of d_out.
// ---------------------------------------------------------------------------
__global__ __launch_bounds__(256) void qk_kernel(
    const __bf16* __restrict__ Qb, const __bf16* __restrict__ Kb,
    float* __restrict__ Eout) {
  const int lane = threadIdx.x & 31;
  const int wave = threadIdx.x >> 5;
  const int t = blockIdx.x * 8 + wave;          // B*64*64 = 32768 wave-tiles
  const int b  = t >> 12;
  const int qt = (t >> 6) & 63;
  const int kt = t & 63;

  const int lcol = lane & 15;
  const int kb   = (lane & 16) ? 8 : 0;
  const int qbase = qt * 32, kbase = kt * 32;

  const __bf16* qp0 = Qb + ((size_t)b * L + qbase + lcol) * D + kb;
  const __bf16* qp1 = qp0 + (size_t)16 * D;
  const __bf16* kp0 = Kb + ((size_t)b * L + kbase + lcol) * D + kb;
  const __bf16* kp1 = kp0 + (size_t)16 * D;

  v8f acc00 = {}, acc01 = {}, acc10 = {}, acc11 = {};
#pragma unroll 2
  for (int d0 = 0; d0 < D; d0 += 32) {
    __builtin_prefetch(qp0 + d0 + 128, 0, 0);
    __builtin_prefetch(kp0 + d0 + 128, 0, 0);
    v16bf a0 = frag_ld(qp0 + d0);
    v16bf a1 = frag_ld(qp1 + d0);
    v16bf b0 = frag_ld(kp0 + d0);
    v16bf b1 = frag_ld(kp1 + d0);
    acc00 = wmma_bf16(a0, b0, acc00);
    acc01 = wmma_bf16(a0, b1, acc01);
    acc10 = wmma_bf16(a1, b0, acc10);
    acc11 = wmma_bf16(a1, b1, acc11);
  }

  const int moff = (lane & 16) ? 8 : 0;
  float* e00 = Eout + ((size_t)b * L + qbase + moff) * L + kbase + lcol;
#pragma unroll
  for (int r = 0; r < 8; ++r) {
    e00[(size_t)r * L]                       = acc00[r] * SCALE_E;
    e00[(size_t)r * L + 16]                  = acc01[r] * SCALE_E;
    e00[(size_t)(r + 16) * L]                = acc10[r] * SCALE_E;
    e00[(size_t)(r + 16) * L + 16]           = acc11[r] * SCALE_E;
  }
}

// ---------------------------------------------------------------------------
// Kernel 3: row softmax. One 256-thread block per (b,q) row of 2048.
// In-place on d_out (f32) + bf16 copy to ws for the A*V GEMM.
// ---------------------------------------------------------------------------
__global__ __launch_bounds__(256) void softmax_kernel(
    float* __restrict__ A, __bf16* __restrict__ Ab) {
  __shared__ float red[256];
  const int row = blockIdx.x;                   // b*L + q
  const int tid = threadIdx.x;
  float*   p  = A  + (size_t)row * L;
  __bf16*  pb = Ab + (size_t)row * L;

  float v[8];
  float m = -3.402823466e38f;
#pragma unroll
  for (int i = 0; i < 8; ++i) { v[i] = p[tid + i * 256]; m = fmaxf(m, v[i]); }
  red[tid] = m; __syncthreads();
  for (int s = 128; s > 0; s >>= 1) {
    if (tid < s) red[tid] = fmaxf(red[tid], red[tid + s]);
    __syncthreads();
  }
  m = red[0]; __syncthreads();

  float sum = 0.f;
#pragma unroll
  for (int i = 0; i < 8; ++i) { v[i] = __expf(v[i] - m); sum += v[i]; }
  red[tid] = sum; __syncthreads();
  for (int s = 128; s > 0; s >>= 1) {
    if (tid < s) red[tid] += red[tid + s];
    __syncthreads();
  }
  const float inv = 1.f / red[0];
#pragma unroll
  for (int i = 0; i < 8; ++i) {
    float o = v[i] * inv;
    p[tid + i * 256]  = o;
    pb[tid + i * 256] = (__bf16)o;
  }
}

// ---------------------------------------------------------------------------
// Kernel 4: C = A V via V^T [b,d,l]. One wave per 32x32 tile (2x2 WMMA):
// both operands now contiguous over k -> all-b128 fragment loads.
// B-frag: B[k][n] = V[k0+k, dbase+n] = Vt[b, dbase+n, k0+k]
// ---------------------------------------------------------------------------
__global__ __launch_bounds__(256) void av_kernel(
    const __bf16* __restrict__ Ab, const __bf16* __restrict__ Vt,
    float* __restrict__ Cout) {
  const int lane = threadIdx.x & 31;
  const int wave = threadIdx.x >> 5;
  const int t = blockIdx.x * 8 + wave;          // B*64*16 = 8192 wave-tiles
  const int b  = t >> 10;
  const int qt = (t >> 4) & 63;
  const int dt = t & 15;

  const int lcol = lane & 15;
  const int kb   = (lane & 16) ? 8 : 0;
  const int qbase = qt * 32, dbase = dt * 32;

  const __bf16* ap0 = Ab + ((size_t)b * L + qbase + lcol) * L + kb;
  const __bf16* ap1 = ap0 + (size_t)16 * L;
  const __bf16* vp0 = Vt + ((size_t)b * D + dbase + lcol) * L + kb;
  const __bf16* vp1 = vp0 + (size_t)16 * L;

  v8f acc00 = {}, acc01 = {}, acc10 = {}, acc11 = {};
#pragma unroll 2
  for (int k0 = 0; k0 < L; k0 += 32) {
    __builtin_prefetch(ap0 + k0 + 128, 0, 0);
    __builtin_prefetch(vp0 + k0 + 128, 0, 0);
    v16bf a0 = frag_ld(ap0 + k0);
    v16bf a1 = frag_ld(ap1 + k0);
    v16bf b0 = frag_ld(vp0 + k0);
    v16bf b1 = frag_ld(vp1 + k0);
    acc00 = wmma_bf16(a0, b0, acc00);
    acc01 = wmma_bf16(a0, b1, acc01);
    acc10 = wmma_bf16(a1, b0, acc10);
    acc11 = wmma_bf16(a1, b1, acc11);
  }

  const int moff = (lane & 16) ? 8 : 0;
  float* c00 = Cout + ((size_t)b * L + qbase + moff) * D + dbase + lcol;
#pragma unroll
  for (int r = 0; r < 8; ++r) {
    c00[(size_t)r * D]             = acc00[r];
    c00[(size_t)r * D + 16]        = acc01[r];
    c00[(size_t)(r + 16) * D]      = acc10[r];
    c00[(size_t)(r + 16) * D + 16] = acc11[r];
  }
}

// ---------------------------------------------------------------------------
extern "C" void kernel_launch(void* const* d_in, const int* in_sizes, int n_in,
                              void* d_out, int out_size, void* d_ws, size_t ws_size,
                              hipStream_t stream) {
  (void)in_sizes; (void)n_in; (void)out_size; (void)ws_size;
  const float* H  = (const float*)d_in[0];
  const float* Wk = (const float*)d_in[1];
  const float* bk = (const float*)d_in[2];
  const float* Wq = (const float*)d_in[3];
  const float* bq = (const float*)d_in[4];
  const float* Wv = (const float*)d_in[5];
  const float* bv = (const float*)d_in[6];

  float* Cout = (float*)d_out;          // [B,L,D] f32
  float* Aout = Cout + BLD;             // [B,L,L] f32

  __bf16* ws  = (__bf16*)d_ws;
  __bf16* KQV = ws;                     // K @ 0, Q @ BLD, V^T @ 2*BLD
  __bf16* Kb  = ws;
  __bf16* Qb  = ws + BLD;
  __bf16* Vt  = ws + 2 * BLD;           // transposed V: [b, d, l]
  __bf16* Ab  = ws + 3 * BLD;           // bf16 softmax output, BLL elems

  proj_kernel<<<(3 * B * 128 * 32) / 8, 256, 0, stream>>>(H, Wk, bk, Wq, bq, Wv, bv, KQV);
  qk_kernel<<<(B * 64 * 64) / 8, 256, 0, stream>>>(Qb, Kb, Aout);
  softmax_kernel<<<B * L, 256, 0, stream>>>(Aout, Ab);
  av_kernel<<<(B * 64 * 16) / 8, 256, 0, stream>>>(Ab, Vt, Cout);
}
